// ProjectedQuantumKernel_81003083202739
// MI455X (gfx1250) — compile-verified, matrix operand order checked by reading, not run on hardware
//
#include <hip/hip_runtime.h>
#include <hip/hip_bf16.h>

// ---------------------------------------------------------------------------
// Reference math summary:
//   probs = one-hot at index 0 (independent of x) -> gathered[n,q] = probs[1<<q]
//   p[n,j] = sum_q gathered[n,q] * basis[j,q,0]          (8192 x 64)
//   pn     = p / (||p||_row + 1e-8)
//   out    = p1n @ p2n^T                                  (8192 x 8192 f32)
// Output write (256 MB) dominates: ~11us at 23.3 TB/s. GEMM K=64 is tiny, so
// WMMA f16-in/f32-acc is used for the matmul stage (codegen-confirmed builtin).
// ---------------------------------------------------------------------------

#define NROWS   8192
#define NPROJ   64      // K dimension of the final GEMM
#define NQ      8

typedef _Float16 v16h __attribute__((ext_vector_type(16)));
typedef float    v8f  __attribute__((ext_vector_type(8)));
typedef _Float16 h8   __attribute__((ext_vector_type(8)));

// ---------------------------------------------------------------------------
// Stage 1: projection + row normalization -> f16 matrices in workspace.
// One thread per row. rows [0,8192) -> P1, [8192,16384) -> P2.
// ---------------------------------------------------------------------------
__global__ void __launch_bounds__(64)
project_normalize_kernel(const float* __restrict__ basis,
                         _Float16* __restrict__ P1,
                         _Float16* __restrict__ P2) {
    int row = blockIdx.x * blockDim.x + threadIdx.x;
    if (row >= 2 * NROWS) return;
    _Float16* dst = (row < NROWS) ? (P1 + (size_t)row * NPROJ)
                                  : (P2 + (size_t)(row - NROWS) * NPROJ);

    // gathered[q] = probs[1<<q], probs = one-hot at index 0 (per reference)
    float g[NQ];
#pragma unroll
    for (int q = 0; q < NQ; ++q)
        g[q] = ((1u << q) == 0u) ? 1.0f : 0.0f;

    float vals[NPROJ];
    float ss = 0.0f;
#pragma unroll
    for (int j = 0; j < NPROJ; ++j) {
        float s = 0.0f;
#pragma unroll
        for (int q = 0; q < NQ; ++q)
            s += g[q] * basis[j * (NQ * 3) + q * 3 + 0];   // basis[j, q, 0]
        vals[j] = s;
        ss += s * s;
    }
    float inv = 1.0f / (sqrtf(ss) + 1e-8f);
#pragma unroll
    for (int j = 0; j < NPROJ; ++j)
        dst[j] = (_Float16)(vals[j] * inv);
}

// ---------------------------------------------------------------------------
// Stage 2: out = P1n @ P2n^T via V_WMMA_F32_16X16X32_F16 (K = 64 -> 2 WMMAs).
// 128 threads = 4 wave32; each wave computes one 16x16 output tile.
// Fragment layout (ISA 7.12.2, 16-bit A 16x32): lane l -> m = l&15, hi = l>>4;
// elements 0..7 hold K = hi*8 + 0..7, elements 8..15 hold K = 16 + hi*8 + 0..7.
// B (32x16) is loaded symmetrically from rows of P2n (out = P1 * P2^T).
// C/D: VGPR v <-> M = v + 8*hi, N = l&15.
// ---------------------------------------------------------------------------
union Frag { v16h v; h8 h[2]; };

__device__ __forceinline__ void load_row_frags(const _Float16* __restrict__ rowp,
                                               int hi, v16h& f0, v16h& f1) {
    Frag a, b;
    a.h[0] = *(const h8*)(rowp +      hi * 8);   // K  0..7  (+hi*8)
    a.h[1] = *(const h8*)(rowp + 16 + hi * 8);   // K 16..23 (+hi*8)
    b.h[0] = *(const h8*)(rowp + 32 + hi * 8);   // K 32..39 (+hi*8)
    b.h[1] = *(const h8*)(rowp + 48 + hi * 8);   // K 48..55 (+hi*8)
    f0 = a.v;
    f1 = b.v;
}

__global__ void __launch_bounds__(128)
gemm_wmma_kernel(const _Float16* __restrict__ P1,
                 const _Float16* __restrict__ P2,
                 float* __restrict__ out) {
    const int lane  = threadIdx.x & 31;
    const int wave  = threadIdx.x >> 5;
    const int tileM = blockIdx.y;                  // 0..511
    const int tileN = blockIdx.x * 4 + wave;       // 0..511
    const int m     = lane & 15;
    const int hi    = lane >> 4;

    const _Float16* arow = P1 + (size_t)(tileM * 16 + m) * NPROJ;
    const _Float16* brow = P2 + (size_t)(tileN * 16 + m) * NPROJ;

    v16h a0, a1, b0, b1;
    load_row_frags(arow, hi, a0, a1);
    load_row_frags(brow, hi, b0, b1);

    v8f c = {};
    c = __builtin_amdgcn_wmma_f32_16x16x32_f16(false, a0, false, b0,
                                               (short)0, c, false, false);
    c = __builtin_amdgcn_wmma_f32_16x16x32_f16(false, a1, false, b1,
                                               (short)0, c, false, false);

    // D: VGPR v -> row tileM*16 + hi*8 + v, col tileN*16 + m.
    // 256MB stream > 192MB L2 -> non-temporal stores.
    float* o = out + (size_t)(tileM * 16 + hi * 8) * NROWS + (size_t)tileN * 16 + m;
#pragma unroll
    for (int v = 0; v < 8; ++v)
        __builtin_nontemporal_store(c[v], o + (size_t)v * NROWS);
}

// ---------------------------------------------------------------------------
// Launch: inputs are (x1, x2, projection_basis); x1/x2 are never read by the
// reference math (probs is independent of x), so only basis (d_in[2]) is used.
// Workspace: P1n (1MB f16) | P2n (1MB f16).
// ---------------------------------------------------------------------------
extern "C" void kernel_launch(void* const* d_in, const int* in_sizes, int n_in,
                              void* d_out, int out_size, void* d_ws, size_t ws_size,
                              hipStream_t stream) {
    (void)in_sizes; (void)n_in; (void)out_size; (void)ws_size;

    const float* basis = (const float*)d_in[2];
    _Float16* P1 = (_Float16*)d_ws;
    _Float16* P2 = (_Float16*)((char*)d_ws + (size_t)NROWS * NPROJ * sizeof(_Float16));
    float* out = (float*)d_out;

    // Stage 1: 16384 rows, 64 threads/block.
    project_normalize_kernel<<<dim3((2 * NROWS) / 64), dim3(64), 0, stream>>>(basis, P1, P2);

    // Stage 2: 512x512 tiles of 16x16; 4 tiles per 128-thread block.
    gemm_wmma_kernel<<<dim3(512 / 4, 512), dim3(128), 0, stream>>>(P1, P2, out);
}
